// DGCNN_cls_46488726011937
// MI455X (gfx1250) — matrix operand order repaired and checked by reading, not compiled, not run on hardware
//
#include <hip/hip_runtime.h>
#include <hip/hip_bf16.h>
#include <stdint.h>

typedef __attribute__((ext_vector_type(16))) _Float16 v16h;
typedef __attribute__((ext_vector_type(8)))  float    v8f;

#define BATCH 16
#define NPTS  1024
#define KNN   20
#define HCH   512   // concat channel count in hbuf

union Frag { uint32_t u[8]; v16h h; };

// ---- monotone float <-> uint map so unsigned atomicMax == float max ----
__device__ __forceinline__ unsigned fmap(float x) {
    unsigned u = __float_as_uint(x);
    return (u & 0x80000000u) ? ~u : (u | 0x80000000u);
}
__device__ __forceinline__ float funmap(unsigned u) {
    return (u & 0x80000000u) ? __uint_as_float(u & 0x7FFFFFFFu)
                             : __uint_as_float(~u);
}

// ---------------------------------------------------------------- utilities
__global__ void zero_u32(unsigned* p, int n) {
    int t = blockIdx.x * blockDim.x + threadIdx.x;
    if (t < n) p[t] = 0u;
}

// x (B,3,N) -> xt (B,N,3)
__global__ void transpose_x(const float* __restrict__ x, float* __restrict__ xt) {
    int t = blockIdx.x * blockDim.x + threadIdx.x;           // over B*N*3
    if (t >= BATCH * NPTS * 3) return;
    int c = t % 3, n = (t / 3) % NPTS, b = t / (3 * NPTS);
    xt[t] = x[(b * 3 + c) * NPTS + n];
}

// ---------------------------------------------------------------- kNN
// feat: base pointer already offset to channel slice; row stride in floats.
// One wave (32 threads) per 32 query points; query tile staged in LDS.
__global__ void knn_kernel(const float* __restrict__ feat, int rowStride, int C,
                           int* __restrict__ idxOut) {
    __shared__ float xq[32][128];
    int b    = blockIdx.x >> 5;          // 32 tiles per batch
    int tile = blockIdx.x & 31;
    int i    = tile * 32 + threadIdx.x;
    // stage 32 query rows
    for (int e = threadIdx.x; e < 32 * C; e += 32) {
        int r = e / C, c = e % C;
        xq[r][c] = feat[((size_t)b * NPTS + tile * 32 + r) * rowStride + c];
    }
    __syncthreads();

    float tv[KNN]; int ti[KNN];
#pragma unroll
    for (int t = 0; t < KNN; ++t) { tv[t] = -1e30f; ti[t] = 0; }

    const float* fb = feat + (size_t)b * NPTS * rowStride;
    for (int j = 0; j < NPTS; ++j) {
        const float* xj = fb + (size_t)j * rowStride;
        float d = 0.f;
        for (int c = 0; c < C; ++c) {
            float df = xq[threadIdx.x][c] - xj[c];
            d -= df * df;                                    // pd = -||xi-xj||^2
        }
        int m = 0; float mv = tv[0];
#pragma unroll
        for (int t = 1; t < KNN; ++t) if (tv[t] < mv) { mv = tv[t]; m = t; }
        if (d > mv) { tv[m] = d; ti[m] = j; }
    }
#pragma unroll
    for (int t = 0; t < KNN; ++t)
        idxOut[((size_t)b * NPTS + i) * KNN + t] = ti[t];
}

// ---------------------------------------------------------------- WMMA GEMM
// Block = 128 threads (4 waves); block tile = 64 rows x 64 output cols.
// Wave w owns columns [w*16, w*16+16); it runs 4 WMMAs per K-chunk (one per
// 16-row M-tile), reusing one B fragment from registers.
// mode 0 (edge conv): rows = (point, k-neighbor) pairs within one batch.
//   A[r][c] = c<Cin ? feat[nb][c]-feat[pt][c] : feat[pt][c-Cin]   (Kdim = 2*Cin)
//   epilogue: BN + LeakyReLU, max over k via atomicMax(umap) into uout(B,N,O)
// mode 1 (embedding conv): rows = (b,n) over all batches.
//   epilogue: BN + LeakyReLU, global max (atomicMax umap) + mean (atomicAdd/N)
__global__ void gemm_conv(const float* __restrict__ feat, int rowStride, int Cin,
                          int Kdim, const int* __restrict__ idx,
                          const float* __restrict__ W,
                          const float* __restrict__ bng, const float* __restrict__ bnb,
                          const float* __restrict__ bnm, const float* __restrict__ bnv,
                          int O, int mode,
                          unsigned* __restrict__ uout, float* __restrict__ gsum) {
    __shared__ _Float16 lA[64][32];
    __shared__ _Float16 lB[64][32];

    const int tid  = threadIdx.x;
    const int wave = tid >> 5, lane = tid & 31;
    const int half = lane >> 4, l15 = lane & 15;

    int b = 0, rowBase;
    if (mode == 0) {
        b       = blockIdx.x / 320;             // N*KNN/64 = 320 block-tiles/batch
        rowBase = (blockIdx.x % 320) * 64;      // row within batch (0..20479)
    } else {
        rowBase = blockIdx.x * 64;              // global row over B*N
    }
    const int obase = blockIdx.y * 64;
    const int KC = (Kdim + 31) / 32;

    v8f acc[4] = {};
    for (int kc = 0; kc < KC; ++kc) {
        const int k0 = kc * 32;
        // ---- stage A tile (64x32): on-the-fly edge-feature gather ----
        for (int e = tid; e < 2048; e += 128) {
            int r = e >> 5, c = e & 31;
            int kg = k0 + c;
            float v = 0.f;
            if (kg < Kdim) {
                if (mode == 0) {
                    int gr = rowBase + r;
                    int pt = gr / KNN, kk = gr % KNN;
                    int nb = idx[((size_t)b * NPTS + pt) * KNN + kk];
                    const float* fb = feat + (size_t)b * NPTS * rowStride;
                    if (kg < Cin) v = fb[(size_t)nb * rowStride + kg] -
                                      fb[(size_t)pt * rowStride + kg];
                    else          v = fb[(size_t)pt * rowStride + (kg - Cin)];
                } else {
                    int gr = rowBase + r;
                    v = feat[(size_t)gr * rowStride + kg];
                }
            }
            lA[r][c] = (_Float16)v;
        }
        // ---- stage B tile (64x32), o-major so K pairs are contiguous ----
        for (int e = tid; e < 2048; e += 128) {
            int o = e >> 5, c = e & 31;
            int kg = k0 + c;
            float v = (kg < Kdim) ? W[(size_t)(obase + o) * Kdim + kg] : 0.f;
            lB[o][c] = (_Float16)v;
        }
        // prefetch next K-chunk of the W panel (global_prefetch_b8)
        if (k0 + 32 < Kdim)
            __builtin_prefetch(&W[(size_t)(obase + (tid >> 1)) * Kdim + k0 + 32], 0, 1);
        __syncthreads();

        // ---- B fragment once per chunk (documented CDNA5 16-bit layout) ----
        Frag bf;
#pragma unroll
        for (int v = 0; v < 8; ++v) {
            int kb = ((v < 4) ? 2 * v : 16 + 2 * (v - 4)) + 8 * half;  // B interleave
            bf.u[v] = *(const uint32_t*)&lB[wave * 16 + l15][kb];
        }
        // ---- 4 M-tiles, one WMMA each, reusing bf ----
#pragma unroll
        for (int mt = 0; mt < 4; ++mt) {
            Frag a;
#pragma unroll
            for (int v = 0; v < 8; ++v) {
                int ka = ((v < 4) ? 2 * v : 16 + 2 * (v - 4)) + 8 * half;  // A interleave
                a.u[v] = *(const uint32_t*)&lA[mt * 16 + l15][ka];
            }
            acc[mt] = __builtin_amdgcn_wmma_f32_16x16x32_f16(
                false, a.h, false, bf.h, (short)0, acc[mt], false, false);
        }
        __syncthreads();
    }

    // ---- epilogue: BN + LeakyReLU + atomic reductions ----
    const int ocol = obase + wave * 16 + l15;
    const float g  = bng[ocol], bt = bnb[ocol], mn = bnm[ocol];
    const float iv = rsqrtf(bnv[ocol] + 1e-5f);
#pragma unroll
    for (int mt = 0; mt < 4; ++mt) {
#pragma unroll
        for (int v = 0; v < 8; ++v) {
            int m   = mt * 16 + v + 8 * half;       // row within block tile
            float y = g * (acc[mt][v] - mn) * iv + bt;
            y = (y >= 0.f) ? y : 0.2f * y;
            int gr = rowBase + m;
            if (mode == 0) {
                int pt = gr / KNN;
                atomicMax(&uout[((size_t)b * NPTS + pt) * O + ocol], fmap(y));
            } else {
                int bb = gr >> 10;                  // N = 1024
                atomicMax(&uout[(size_t)bb * O + ocol], fmap(y));
                atomicAdd(&gsum[(size_t)bb * O + ocol], y * (1.0f / NPTS));
            }
        }
    }
}

// unmap u32-max buffer into hbuf channel slice
__global__ void unmap_kernel(const unsigned* __restrict__ u, float* __restrict__ hbuf,
                             int O, int coff) {
    int t = blockIdx.x * blockDim.x + threadIdx.x;       // over B*N*O
    if (t >= BATCH * NPTS * O) return;
    int o = t % O, r = t / O;                            // r = b*N+n
    hbuf[(size_t)r * HCH + coff + o] = funmap(u[t]);
}

// g[b][0..1023] = max, g[b][1024..2047] = mean
__global__ void finalize_g(const unsigned* __restrict__ gmax,
                           const float* __restrict__ gsum, float* __restrict__ g) {
    int t = blockIdx.x * blockDim.x + threadIdx.x;       // over B*1024
    if (t >= BATCH * 1024) return;
    int b = t >> 10, o = t & 1023;
    g[(size_t)b * 2048 + o]        = funmap(gmax[t]);
    g[(size_t)b * 2048 + 1024 + o] = gsum[t];
}

// FC: out[b][o] = act(bn(in[b]·W[o] + bias))
__global__ void fc_kernel(const float* __restrict__ in, const float* __restrict__ W,
                          const float* __restrict__ bias,
                          const float* __restrict__ bng, const float* __restrict__ bnb,
                          const float* __restrict__ bnm, const float* __restrict__ bnv,
                          int Kd, int Od, int do_lrelu, float* __restrict__ out) {
    int t = blockIdx.x * blockDim.x + threadIdx.x;
    if (t >= BATCH * Od) return;
    int b = t / Od, o = t % Od;
    float s = 0.f;
    for (int c = 0; c < Kd; ++c) s += in[(size_t)b * Kd + c] * W[(size_t)o * Kd + c];
    if (bias) s += bias[o];
    if (bng)  s = bng[o] * (s - bnm[o]) * rsqrtf(bnv[o] + 1e-5f) + bnb[o];
    if (do_lrelu) s = (s >= 0.f) ? s : 0.2f * s;
    out[t] = s;
}

// ---------------------------------------------------------------- launch
extern "C" void kernel_launch(void* const* d_in, const int* in_sizes, int n_in,
                              void* d_out, int out_size, void* d_ws, size_t ws_size,
                              hipStream_t stream) {
    (void)in_sizes; (void)n_in; (void)out_size; (void)ws_size;
    // input order: x, W1, bn1(g,b,m,v), W2, bn2, W3, bn3, W4, bn4, W5, bn5,
    //              L1, bn6, L2, b2, bn7, L3, b3
    const float* x  = (const float*)d_in[0];
    const float* W1 = (const float*)d_in[1];
    const float* const* bn1 = (const float* const*)&d_in[2];
    const float* W2 = (const float*)d_in[6];
    const float* const* bn2 = (const float* const*)&d_in[7];
    const float* W3 = (const float*)d_in[11];
    const float* const* bn3 = (const float* const*)&d_in[12];
    const float* W4 = (const float*)d_in[16];
    const float* const* bn4 = (const float* const*)&d_in[17];
    const float* W5 = (const float*)d_in[21];
    const float* const* bn5 = (const float* const*)&d_in[22];
    const float* L1 = (const float*)d_in[26];
    const float* const* bn6 = (const float* const*)&d_in[27];
    const float* L2 = (const float*)d_in[31];
    const float* b2 = (const float*)d_in[32];
    const float* const* bn7 = (const float* const*)&d_in[33];
    const float* L3 = (const float*)d_in[37];
    const float* b3 = (const float*)d_in[38];

    char* ws = (char*)d_ws;
    size_t off = 0;
    float*    xt   = (float*)(ws + off);    off += (size_t)BATCH * NPTS * 3 * 4;
    int*      idx  = (int*)(ws + off);      off += (size_t)BATCH * NPTS * KNN * 4;
    float*    hbuf = (float*)(ws + off);    off += (size_t)BATCH * NPTS * HCH * 4;
    unsigned* ubuf = (unsigned*)(ws + off); off += (size_t)BATCH * NPTS * 256 * 4;
    unsigned* gmax = (unsigned*)(ws + off); off += (size_t)BATCH * 1024 * 4;
    float*    gsum = (float*)(ws + off);    off += (size_t)BATCH * 1024 * 4;
    float*    gvec = (float*)(ws + off);    off += (size_t)BATCH * 2048 * 4;
    float*    z1   = (float*)(ws + off);    off += (size_t)BATCH * 512 * 4;
    float*    z2   = (float*)(ws + off);    off += (size_t)BATCH * 256 * 4;

    const int MT_EDGE = BATCH * (NPTS * KNN / 64);   // 5120 block tiles (64 rows)
    const int MT_EMB  = (BATCH * NPTS) / 64;         // 256 block tiles

    // 0. transpose x -> (B,N,3)
    transpose_x<<<(BATCH * NPTS * 3 + 255) / 256, 256, 0, stream>>>(x, xt);

    struct Layer { const float* feat; int rs, Cin, O, coff;
                   const float* W; const float* const* bn; };
    Layer layers[4] = {
        { xt,   3,   3,   64,  0,   W1, bn1 },
        { hbuf, HCH, 64,  64,  64,  W2, bn2 },
        { hbuf + 64,  HCH, 64,  128, 128, W3, bn3 },
        { hbuf + 128, HCH, 128, 256, 256, W4, bn4 },
    };

    for (int l = 0; l < 4; ++l) {
        Layer& L = layers[l];
        // kNN on current features
        knn_kernel<<<BATCH * 32, 32, 0, stream>>>(L.feat, L.rs, L.Cin, idx);
        // clear max accumulator
        int ucnt = BATCH * NPTS * L.O;
        zero_u32<<<(ucnt + 255) / 256, 256, 0, stream>>>(ubuf, ucnt);
        // fused gather + WMMA GEMM + BN/LReLU + k-max
        dim3 grid(MT_EDGE, L.O / 64);
        gemm_conv<<<grid, 128, 0, stream>>>(L.feat, L.rs, L.Cin, 2 * L.Cin, idx,
                                            L.W, L.bn[0], L.bn[1], L.bn[2], L.bn[3],
                                            L.O, 0, ubuf, nullptr);
        // unmap into hbuf channel slice
        unmap_kernel<<<(ucnt + 255) / 256, 256, 0, stream>>>(ubuf, hbuf, L.O, L.coff);
    }

    // embedding conv (512 -> 1024) with fused global max+mean pooling
    zero_u32<<<(BATCH * 1024 + 255) / 256, 256, 0, stream>>>(gmax, BATCH * 1024);
    zero_u32<<<(BATCH * 1024 + 255) / 256, 256, 0, stream>>>((unsigned*)gsum, BATCH * 1024);
    {
        dim3 grid(MT_EMB, 1024 / 64);
        gemm_conv<<<grid, 128, 0, stream>>>(hbuf, HCH, HCH, HCH, nullptr,
                                            W5, bn5[0], bn5[1], bn5[2], bn5[3],
                                            1024, 1, gmax, gsum);
    }
    finalize_g<<<(BATCH * 1024 + 255) / 256, 256, 0, stream>>>(gmax, gsum, gvec);

    // FC head
    fc_kernel<<<(BATCH * 512 + 255) / 256, 256, 0, stream>>>(
        gvec, L1, nullptr, bn6[0], bn6[1], bn6[2], bn6[3], 2048, 512, 1, z1);
    fc_kernel<<<(BATCH * 256 + 255) / 256, 256, 0, stream>>>(
        z1, L2, b2, bn7[0], bn7[1], bn7[2], bn7[3], 512, 256, 1, z2);
    fc_kernel<<<(BATCH * 40 + 255) / 256, 256, 0, stream>>>(
        z2, L3, b3, nullptr, nullptr, nullptr, nullptr, 256, 40, 0, (float*)d_out);
}
